// GatedSparseAttention_23149873725783
// MI455X (gfx1250) — matrix operand order repaired and checked by reading, not compiled, not run on hardware
//
#include <hip/hip_runtime.h>

// ---------------------------------------------------------------------------
// GatedSparseAttention for MI455X (gfx1250, wave32, WMMA)
//   E=1024, H=16, D=64, B=2, S=2048
//   GEMMs: pre-split bf16x3 on v_wmma_f32_16x16x32_bf16 (f32 accumulate),
//          A panel staged in LDS via GLOBAL_LOAD_ASYNC_TO_LDS_B128 (ASYNCcnt)
//   Attention: sparse (<=6 keys/row) lane-per-row kernel + cooperative row-0
// ---------------------------------------------------------------------------

typedef __attribute__((ext_vector_type(4)))  __bf16 v4bf;
typedef __attribute__((ext_vector_type(8)))  __bf16 v8bf;
typedef __attribute__((ext_vector_type(16))) __bf16 v16bf;
typedef __attribute__((ext_vector_type(8)))  float  v8f;

#define EDIM 1024
#define SLEN 2048
#define NH   16
#define HD   64
#define MROWS 4096            // B * S
#define TWO_PI_OVER_P 0.0156687912890890f   // 2*pi/401
#define NEGINF (-__builtin_inff())

__device__ __forceinline__ __bf16 bf16_hi(float f) { return (__bf16)f; }
__device__ __forceinline__ __bf16 bf16_lo(float f, __bf16 h)
{
    return (__bf16)(f - (float)h);
}
__device__ __forceinline__ void bf16_split(float f, __bf16& h, __bf16& l)
{
    h = bf16_hi(f);
    l = bf16_lo(f, h);
}

// ---------------------------------------------------------------------------
// Pre-pass: split f32 array into bf16 hi + bf16 lo arrays (4 elems / thread)
// ---------------------------------------------------------------------------
__global__ __launch_bounds__(256) void gsa_split(
    const float* __restrict__ in, __bf16* __restrict__ hi,
    __bf16* __restrict__ lo, int n4)
{
    const int i = blockIdx.x * 256 + threadIdx.x;
    if (i >= n4) return;
    const float4 f = ((const float4*)in)[i];
    v4bf hv, lv;
    __bf16 h;
    h = bf16_hi(f.x); hv[0] = h; lv[0] = bf16_lo(f.x, h);
    h = bf16_hi(f.y); hv[1] = h; lv[1] = bf16_lo(f.y, h);
    h = bf16_hi(f.z); hv[2] = h; lv[2] = bf16_lo(f.z, h);
    h = bf16_hi(f.w); hv[3] = h; lv[3] = bf16_lo(f.w, h);
    ((v4bf*)hi)[i] = hv;
    ((v4bf*)lo)[i] = lv;
}

// ---------------------------------------------------------------------------
// GEMM: out = A[M,EDIM] @ W[n,k]^T + bias[n], operands pre-split bf16 hi/lo.
//   MODE 0: out row-major [M, EDIM]              (output projection)
//   MODE 1: out scattered to [B, H, S, D] layout (Q/K/V projections)
// Block = 8 waves covering a 16(M) x 128(N) tile. The block's 16-row A panel
// (contiguous 32KB per hi/lo) is staged into LDS with async copies; each wave
// then computes one 16x16 tile, 3 WMMAs per 32-wide K step (hi*hi+hi*lo+lo*hi).
// ---------------------------------------------------------------------------
template <int MODE>
__global__ __launch_bounds__(256) void gsa_gemm_bf16x3(
    const __bf16* __restrict__ Ahi, const __bf16* __restrict__ Alo,
    const __bf16* __restrict__ Whi, const __bf16* __restrict__ Wlo,
    const float* __restrict__ bias, float* __restrict__ out)
{
    __shared__ __bf16 sAhi[16 * EDIM];   // 32 KB
    __shared__ __bf16 sAlo[16 * EDIM];   // 32 KB

    const int tid  = threadIdx.x;
    const int lane = tid & 31;
    const int wave = tid >> 5;
    const int half = lane >> 4;       // 0: lanes 0-15, 1: lanes 16-31
    const int l15  = lane & 15;

    const int n0 = (blockIdx.x * 8 + wave) * 16;   // output col tile
    const int m0 = blockIdx.y * 16;                // output row tile
    const int n  = n0 + l15;

    // ---- async-stage the contiguous 32KB hi/lo A panels into LDS ----------
    {
        const unsigned long long ghi =
            (unsigned long long)(const void*)(Ahi + (size_t)m0 * EDIM);
        const unsigned long long glo =
            (unsigned long long)(const void*)(Alo + (size_t)m0 * EDIM);
        const unsigned lbhi = (unsigned)(unsigned long long)(const void*)&sAhi[0];
        const unsigned lblo = (unsigned)(unsigned long long)(const void*)&sAlo[0];
#pragma unroll
        for (int u = 0; u < 8; ++u) {
            const unsigned off = (unsigned)(tid + u * 256) * 16u;  // 16B chunks
            asm volatile("global_load_async_to_lds_b128 %0, %1, off"
                         :: "v"(lbhi + off), "v"(ghi + (unsigned long long)off)
                         : "memory");
            asm volatile("global_load_async_to_lds_b128 %0, %1, off"
                         :: "v"(lblo + off), "v"(glo + (unsigned long long)off)
                         : "memory");
        }
        asm volatile("s_wait_asynccnt 0x0" ::: "memory");
    }
    __syncthreads();

    const __bf16* __restrict__ sah = sAhi + l15 * EDIM;  // lane's A row
    const __bf16* __restrict__ sal = sAlo + l15 * EDIM;
    const __bf16* __restrict__ wh  = Whi + (size_t)n * EDIM;
    const __bf16* __restrict__ wl  = Wlo + (size_t)n * EDIM;

    v8f acc = {};

    for (int k0 = 0; k0 < EDIM; k0 += 32) {
        // A fragment (16x32): lane holds 16 contiguous K [c0..c0+15]
        const int c0 = k0 + half * 16;
        const int c1 = c0 + 8;
        const v8bf ah0 = *(const v8bf*)(sah + c0);
        const v8bf ah1 = *(const v8bf*)(sah + c1);
        const v8bf al0 = *(const v8bf*)(sal + c0);
        const v8bf al1 = *(const v8bf*)(sal + c1);
        // B fragment (32x16): lane holds 16 contiguous K of column n
        const int cB = k0 + half * 16;
        const v8bf bh0 = *(const v8bf*)(wh + cB);
        const v8bf bh1 = *(const v8bf*)(wh + cB + 8);
        const v8bf bl0 = *(const v8bf*)(wl + cB);
        const v8bf bl1 = *(const v8bf*)(wl + cB + 8);

        const v16bf fahi = __builtin_shufflevector(ah0, ah1,
            0,1,2,3,4,5,6,7,8,9,10,11,12,13,14,15);
        const v16bf falo = __builtin_shufflevector(al0, al1,
            0,1,2,3,4,5,6,7,8,9,10,11,12,13,14,15);
        const v16bf fbhi = __builtin_shufflevector(bh0, bh1,
            0,1,2,3,4,5,6,7,8,9,10,11,12,13,14,15);
        const v16bf fblo = __builtin_shufflevector(bl0, bl1,
            0,1,2,3,4,5,6,7,8,9,10,11,12,13,14,15);

        acc = __builtin_amdgcn_wmma_f32_16x16x32_bf16(false, fahi, false, fbhi,
                                                      (short)0, acc, false, false);
        acc = __builtin_amdgcn_wmma_f32_16x16x32_bf16(false, fahi, false, fblo,
                                                      (short)0, acc, false, false);
        acc = __builtin_amdgcn_wmma_f32_16x16x32_bf16(false, falo, false, fbhi,
                                                      (short)0, acc, false, false);
    }

    const float bn = bias[n];
#pragma unroll
    for (int j = 0; j < 8; ++j) {
        const int mm  = m0 + j + 8 * half;   // C/D layout: M = j (+8 hi lanes)
        const float v = acc[j] + bn;
        if (MODE == 0) {
            out[(size_t)mm * EDIM + n] = v;
        } else {
            const int b = mm >> 11, s = mm & (SLEN - 1);
            const int h = n >> 6,   d = n & (HD - 1);
            out[(((size_t)(b * NH + h) * SLEN) + s) * HD + d] = v;
        }
    }
}

// ---------------------------------------------------------------------------
// vmean[b,h,d] = mean_j V[b,h,j,d]  (fallback for fully-gated rows)
// ---------------------------------------------------------------------------
__global__ void gsa_vmean(const float* __restrict__ V, float* __restrict__ vmean)
{
    const int bh = blockIdx.x;
    const int d  = threadIdx.x;
    const float* vp = V + (size_t)bh * SLEN * HD + d;
    float s = 0.f;
    for (int j = 0; j < SLEN; ++j) s += vp[(size_t)j * HD];
    vmean[bh * HD + d] = s * (1.0f / SLEN);
}

// ---------------------------------------------------------------------------
// Row i=0 (global query): dense softmax over all S keys. One block per (b,h).
// LDS tree reductions for max/count, ds_add_f32 shared atomics for context.
// Writes ctx already split into bf16 hi/lo for the output-projection GEMM.
// ---------------------------------------------------------------------------
__global__ __launch_bounds__(256) void gsa_attn_row0(
    const float* __restrict__ Q, const float* __restrict__ K,
    const float* __restrict__ V, const float* __restrict__ vmean,
    __bf16* __restrict__ Chi, __bf16* __restrict__ Clo)
{
    __shared__ float sred[256];
    __shared__ int   scnt[256];
    __shared__ float sctx[HD];
    __shared__ float sden;

    const int bh  = blockIdx.x;
    const int tid = threadIdx.x;

    const float* q = Q + (size_t)bh * SLEN * HD;   // q row (i = 0)
    float qv[HD];
#pragma unroll
    for (int t = 0; t < 16; ++t) {
        float4 f = ((const float4*)q)[t];
        qv[4*t+0] = f.x; qv[4*t+1] = f.y; qv[4*t+2] = f.z; qv[4*t+3] = f.w;
    }

    float sc[8];
    float lmax = NEGINF;
    int   lcnt = 0;
#pragma unroll
    for (int u = 0; u < 8; ++u) {
        const int j = tid + u * 256;
        const float* kr = K + ((size_t)bh * SLEN + j) * HD;
        float dot = 0.f;
#pragma unroll
        for (int t = 0; t < 16; ++t) {
            float4 kf = ((const float4*)kr)[t];
            dot = fmaf(qv[4*t+0], kf.x, dot);
            dot = fmaf(qv[4*t+1], kf.y, dot);
            dot = fmaf(qv[4*t+2], kf.z, dot);
            dot = fmaf(qv[4*t+3], kf.w, dot);
        }
        const float diff = (float)(-j);                 // i - j with i = 0
        const float pb   = __expf(-0.125f * diff * diff)
                         + __cosf(TWO_PI_OVER_P * diff);
        const float s = dot * 0.125f + pb;              // 1/sqrt(64) = 0.125
        if (s >= 0.f) { ++lcnt; lmax = fmaxf(lmax, s); sc[u] = s; }
        else          { sc[u] = NEGINF; }               // gated off
    }

    sred[tid] = lmax; scnt[tid] = lcnt;
    __syncthreads();
    for (int off = 128; off > 0; off >>= 1) {
        if (tid < off) {
            sred[tid] = fmaxf(sred[tid], sred[tid + off]);
            scnt[tid] += scnt[tid + off];
        }
        __syncthreads();
    }
    const float gmax = sred[0];
    const int   gcnt = scnt[0];
    __syncthreads();

    if (tid < HD) sctx[tid] = 0.f;
    if (tid == 0) sden = 0.f;
    __syncthreads();

    const int b = bh >> 4, h = bh & 15;
    const size_t coff = (size_t)b * SLEN * EDIM + h * HD;   // row s = 0

    if (gcnt == 0) {   // everything gated: softmax degenerates to uniform 1/S
        if (tid < HD) {
            const float val = vmean[bh * HD + tid];
            __bf16 hh, ll; bf16_split(val, hh, ll);
            Chi[coff + tid] = hh; Clo[coff + tid] = ll;
        }
        return;
    }

    float part[HD];
#pragma unroll
    for (int t = 0; t < HD; ++t) part[t] = 0.f;
    float lden = 0.f;
#pragma unroll
    for (int u = 0; u < 8; ++u) {
        if (sc[u] == NEGINF) continue;
        const int j = tid + u * 256;
        const float p = __expf(sc[u] - gmax);
        lden += p;
        const float* vr = V + ((size_t)bh * SLEN + j) * HD;
#pragma unroll
        for (int t = 0; t < 16; ++t) {
            float4 vf = ((const float4*)vr)[t];
            part[4*t+0] = fmaf(p, vf.x, part[4*t+0]);
            part[4*t+1] = fmaf(p, vf.y, part[4*t+1]);
            part[4*t+2] = fmaf(p, vf.z, part[4*t+2]);
            part[4*t+3] = fmaf(p, vf.w, part[4*t+3]);
        }
    }
    atomicAdd(&sden, lden);
#pragma unroll
    for (int t = 0; t < HD; ++t) atomicAdd(&sctx[t], part[t]);
    __syncthreads();
    if (tid < HD) {
        const float val = sctx[tid] / sden;
        __bf16 hh, ll; bf16_split(val, hh, ll);
        Chi[coff + tid] = hh; Clo[coff + tid] = ll;
    }
}

// ---------------------------------------------------------------------------
// Sparse rows i >= 1: each lane owns one (b,h,i). Candidates = {0} U [i-2,i+2].
// ---------------------------------------------------------------------------
__global__ __launch_bounds__(256) void gsa_attn_sparse(
    const float* __restrict__ Q, const float* __restrict__ K,
    const float* __restrict__ V, const float* __restrict__ vmean,
    __bf16* __restrict__ Chi, __bf16* __restrict__ Clo)
{
    const int r  = blockIdx.x * 256 + threadIdx.x;   // 0 .. B*H*S-1
    const int i  = r & (SLEN - 1);
    const int bh = r >> 11;
    if (i == 0) return;                              // handled by gsa_attn_row0

    const float* q = Q + ((size_t)bh * SLEN + i) * HD;
    float qv[HD];
#pragma unroll
    for (int t = 0; t < 16; ++t) {
        float4 f = ((const float4*)q)[t];
        qv[4*t+0] = f.x; qv[4*t+1] = f.y; qv[4*t+2] = f.z; qv[4*t+3] = f.w;
    }

    const int lo = (i > 2) ? i - 2 : 0;
    const int hi = (i + 2 < SLEN) ? i + 2 : SLEN - 1;

    int  cj[6]; bool cv[6];
    cj[0] = 0;      cv[0] = (lo > 0);        // global token (skip if in window)
#pragma unroll
    for (int t = 0; t < 5; ++t) { cj[t+1] = lo + t; cv[t+1] = (lo + t <= hi); }

    float sc[6];
    float mx = NEGINF;
    int nkeep = 0;
#pragma unroll
    for (int c = 0; c < 6; ++c) {
        sc[c] = NEGINF;
        if (!cv[c]) continue;
        const int j = cj[c];
        const float* kr = K + ((size_t)bh * SLEN + j) * HD;
        float dot = 0.f;
#pragma unroll
        for (int t = 0; t < 16; ++t) {
            float4 kf = ((const float4*)kr)[t];
            dot = fmaf(qv[4*t+0], kf.x, dot);
            dot = fmaf(qv[4*t+1], kf.y, dot);
            dot = fmaf(qv[4*t+2], kf.z, dot);
            dot = fmaf(qv[4*t+3], kf.w, dot);
        }
        const float diff = (float)(i - j);
        const float s = dot * 0.125f
                      + __expf(-0.125f * diff * diff)
                      + __cosf(TWO_PI_OVER_P * diff);
        if (s >= 0.f) { ++nkeep; mx = fmaxf(mx, s); sc[c] = s; }
    }

    const int b = bh >> 4, h = bh & 15;
    const size_t coff = ((size_t)b * SLEN + i) * EDIM + h * HD;
    __bf16* dh = Chi + coff;
    __bf16* dl = Clo + coff;

    if (nkeep == 0) {   // all gated -> uniform over all S -> mean(V)
        const float* vm = vmean + bh * HD;
#pragma unroll
        for (int t = 0; t < 16; ++t) {
            float4 f = ((const float4*)vm)[t];
            v4bf hv, lv;
            __bf16 h;
            h = bf16_hi(f.x); hv[0] = h; lv[0] = bf16_lo(f.x, h);
            h = bf16_hi(f.y); hv[1] = h; lv[1] = bf16_lo(f.y, h);
            h = bf16_hi(f.z); hv[2] = h; lv[2] = bf16_lo(f.z, h);
            h = bf16_hi(f.w); hv[3] = h; lv[3] = bf16_lo(f.w, h);
            ((v4bf*)dh)[t] = hv; ((v4bf*)dl)[t] = lv;
        }
        return;
    }

    float4 acc[16];
#pragma unroll
    for (int t = 0; t < 16; ++t) acc[t] = make_float4(0.f, 0.f, 0.f, 0.f);
    float den = 0.f;
#pragma unroll
    for (int c = 0; c < 6; ++c) {
        if (sc[c] == NEGINF) continue;
        const float p = __expf(sc[c] - mx);
        den += p;
        const float* vr = V + ((size_t)bh * SLEN + cj[c]) * HD;
#pragma unroll
        for (int t = 0; t < 16; ++t) {
            float4 vf = ((const float4*)vr)[t];
            acc[t].x = fmaf(p, vf.x, acc[t].x);
            acc[t].y = fmaf(p, vf.y, acc[t].y);
            acc[t].z = fmaf(p, vf.z, acc[t].z);
            acc[t].w = fmaf(p, vf.w, acc[t].w);
        }
    }
    const float inv = 1.0f / den;
#pragma unroll
    for (int t = 0; t < 16; ++t) {
        v4bf hv, lv;
        __bf16 h;
        const float ox = acc[t].x * inv, oy = acc[t].y * inv;
        const float oz = acc[t].z * inv, ow = acc[t].w * inv;
        h = bf16_hi(ox); hv[0] = h; lv[0] = bf16_lo(ox, h);
        h = bf16_hi(oy); hv[1] = h; lv[1] = bf16_lo(oy, h);
        h = bf16_hi(oz); hv[2] = h; lv[2] = bf16_lo(oz, h);
        h = bf16_hi(ow); hv[3] = h; lv[3] = bf16_lo(ow, h);
        ((v4bf*)dh)[t] = hv; ((v4bf*)dl)[t] = lv;
    }
}

// ---------------------------------------------------------------------------
// Launch: split(x, W*) -> QKV GEMMs -> vmean -> row0 + sparse attn (split ctx)
//         -> output GEMM
// Workspace: Q,K,V f32 (48MB) + vmean (8KB) + x/W/ctx bf16 hi+lo (~48MB)
// ---------------------------------------------------------------------------
extern "C" void kernel_launch(void* const* d_in, const int* in_sizes, int n_in,
                              void* d_out, int out_size, void* d_ws, size_t ws_size,
                              hipStream_t stream)
{
    (void)in_sizes; (void)n_in; (void)out_size; (void)ws_size;
    const float* x  = (const float*)d_in[0];
    const float* wq = (const float*)d_in[1];
    const float* bq = (const float*)d_in[2];
    const float* wk = (const float*)d_in[3];
    const float* bk = (const float*)d_in[4];
    const float* wv = (const float*)d_in[5];
    const float* bv = (const float*)d_in[6];
    const float* wo = (const float*)d_in[7];
    const float* bo = (const float*)d_in[8];
    float* out = (float*)d_out;

    const size_t NX = (size_t)MROWS * EDIM;   // 4M elements (x, ctx)
    const size_t NW = (size_t)EDIM * EDIM;    // 1M elements (each weight)

    float* Qb = (float*)d_ws;                 // [B,H,S,D]
    float* Kb = Qb + NX;
    float* Vb = Kb + NX;
    float* Vm = Vb + NX;                      // [B,H,D] = 2048 floats
    __bf16* Xhi = (__bf16*)(Vm + 2 * NH * HD);
    __bf16* Xlo = Xhi + NX;
    __bf16* Whi[4]; __bf16* Wlo[4];
    {
        __bf16* p = Xlo + NX;
        for (int w = 0; w < 4; ++w) { Whi[w] = p; p += NW; Wlo[w] = p; p += NW; }
    }
    __bf16* Chi = Wlo[3] + NW;
    __bf16* Clo = Chi + NX;

    // ---- split passes ----
    gsa_split<<<dim3((unsigned)(NX / 4 / 256)), dim3(256), 0, stream>>>(
        x, Xhi, Xlo, (int)(NX / 4));
    const float* wsrc[4] = { wq, wk, wv, wo };
    for (int w = 0; w < 4; ++w)
        gsa_split<<<dim3((unsigned)(NW / 4 / 256)), dim3(256), 0, stream>>>(
            wsrc[w], Whi[w], Wlo[w], (int)(NW / 4));

    // ---- QKV projections ----
    const dim3 ggrid(EDIM / 128, MROWS / 16);   // (8, 256)
    const dim3 gblk(256);
    gsa_gemm_bf16x3<1><<<ggrid, gblk, 0, stream>>>(Xhi, Xlo, Whi[0], Wlo[0], bq, Qb);
    gsa_gemm_bf16x3<1><<<ggrid, gblk, 0, stream>>>(Xhi, Xlo, Whi[1], Wlo[1], bk, Kb);
    gsa_gemm_bf16x3<1><<<ggrid, gblk, 0, stream>>>(Xhi, Xlo, Whi[2], Wlo[2], bv, Vb);

    // ---- attention ----
    gsa_vmean<<<dim3(2 * NH), dim3(HD), 0, stream>>>(Vb, Vm);
    gsa_attn_row0<<<dim3(2 * NH), dim3(256), 0, stream>>>(Qb, Kb, Vb, Vm, Chi, Clo);
    gsa_attn_sparse<<<dim3((2 * NH * SLEN) / 256), dim3(256), 0, stream>>>(
        Qb, Kb, Vb, Vm, Chi, Clo);

    // ---- output projection ----
    gsa_gemm_bf16x3<0><<<ggrid, gblk, 0, stream>>>(Chi, Clo, Whi[3], Wlo[3], bo, out);
}
